// OneStepLSTMLayerNorm_44676249813258
// MI455X (gfx1250) — matrix operand order repaired and checked by reading, not compiled, not run on hardware
//
#include <hip/hip_runtime.h>
#include <math.h>

// ---------------------------------------------------------------------------
// Fused LayerNorm-LSTM single step for MI455X (gfx1250, wave32, WMMA).
//   B = 65536, D = 256, U = 256, gates = 4 (N = 1024)
// Design:
//   kernel 1: convert + transpose weights f32 -> bf16 [N][K] into d_ws
//   kernel 2: per block: 32 rows. GEMM phase: v_wmma_f32_16x16x32_bf16,
//             A staged in LDS (bf16), B streamed from L2-resident bf16
//             weights, Wx/Uh accumulated in f32 and stored to LDS.
//             LN/gate phase: wave32 shuffle reductions, fused nonlinearities,
//             h and c written straight to d_out.
// ---------------------------------------------------------------------------

typedef __attribute__((ext_vector_type(16))) __bf16 v16bf;
typedef __attribute__((ext_vector_type(8)))  __bf16 v8bf;
typedef __attribute__((ext_vector_type(8)))  float  v8f;

#define EPS_LN 1e-8f
#define M_TILE 32       // rows per workgroup (two 16-row WMMA subtiles)
#define DK     256      // K dimension (D == U == 256)
#define N4     1024     // 4 gates * 256 units
#define NTHREADS 256    // 8 waves of 32

// LDS layout (dynamic, 288 KB total):
//   [0, 128K)   : ZW  float[32][1024]   (Wx results)
//   [128K,256K) : ZU  float[32][1024]   (Uh results)
//   [256K,288K) : phase1: AX bf16[32][256] + AH bf16[32][256]
//                 phase2: CT float[32][256] (new cell values for LN)
#define SMEM_BYTES (288 * 1024)

__device__ __forceinline__ __bf16 f2bf(float f) {
    unsigned u = __builtin_bit_cast(unsigned, f);
    u += 0x7FFFu + ((u >> 16) & 1u);          // round-to-nearest-even
    unsigned short s = (unsigned short)(u >> 16);
    return __builtin_bit_cast(__bf16, s);
}

__device__ __forceinline__ float wave_sum(float v) {
#pragma unroll
    for (int o = 16; o > 0; o >>= 1) v += __shfl_xor(v, o, 32);
    return v;
}

__device__ __forceinline__ float sigmoidf(float z) {
    return 1.0f / (1.0f + __expf(-z));
}

// A-fragment (16x32 bf16, ISA layout): per lane, two contiguous 8-element
// chunks at +0 and +16 of the lane's K base.
__device__ __forceinline__ v16bf load_a_frag(const __bf16* p) {
    v8bf lo = *(const v8bf*)(p);
    v8bf hi = *(const v8bf*)(p + 16);
    v16bf a;
#pragma unroll
    for (int i = 0; i < 8; ++i) { a[i] = lo[i]; a[i + 8] = hi[i]; }
    return a;
}

// ---------------------------------------------------------------------------
// Kernel 1: weights f32 [K=256][N=1024] -> bf16 transposed [N=1024][K=256]
// ---------------------------------------------------------------------------
__global__ void __launch_bounds__(256)
cvt_weights(const float* __restrict__ Kw, const float* __restrict__ Rw,
            __bf16* __restrict__ Wt, __bf16* __restrict__ Ut) {
    int idx = blockIdx.x * blockDim.x + threadIdx.x;   // 0 .. 256*1024-1
    if (idx < DK * N4) {
        int n = idx >> 8;          // 0..1023
        int k = idx & 255;         // 0..255
        Wt[idx] = f2bf(Kw[k * N4 + n]);
        Ut[idx] = f2bf(Rw[k * N4 + n]);
    }
}

// ---------------------------------------------------------------------------
// Kernel 2: fused GEMM + per-gate LayerNorm + LSTM gates
// ---------------------------------------------------------------------------
__global__ void __launch_bounds__(NTHREADS, 1)
lstm_fused(const float* __restrict__ x, const float* __restrict__ hin,
           const float* __restrict__ cin,
           const __bf16* __restrict__ Wt, const __bf16* __restrict__ Ut,
           const float* __restrict__ g1, const float* __restrict__ b1,
           const float* __restrict__ g2, const float* __restrict__ b2,
           const float* __restrict__ g3, const float* __restrict__ b3,
           float* __restrict__ outH, float* __restrict__ outC) {
    extern __shared__ char smem[];
    float*  ZW = (float*)smem;                       // [32][1024]
    float*  ZU = ZW + M_TILE * N4;                   // [32][1024]
    __bf16* AX = (__bf16*)(smem + 256 * 1024);       // [32][256]
    __bf16* AH = AX + M_TILE * DK;                   // [32][256]
    float*  CT = (float*)(smem + 256 * 1024);        // [32][256] (phase 2)

    const int tid  = threadIdx.x;
    const int wave = tid >> 5;
    const int lane = tid & 31;
    const int row0 = blockIdx.x * M_TILE;

    // ---- stage x/h tiles to LDS as bf16 (coalesced f32 reads) ----
    for (int idx = tid; idx < M_TILE * DK; idx += NTHREADS) {
        int r = idx >> 8, c = idx & 255;
        size_t goff = (size_t)(row0 + r) * DK + c;
        AX[idx] = f2bf(x[goff]);
        AH[idx] = f2bf(hin[goff]);
    }
    __syncthreads();

    // ---- GEMM phase: 128 jobs = {Wx, Uh} x 64 n-tiles of 16 columns ----
    const int lrow = lane & 15;     // row (A) / column (B,C) within tile
    const int hi   = lane >> 4;     // lane-half selector per ISA layout

    for (int j = wave; j < 128; j += 8) {
        const int mat = j & 1;
        const int n0  = (j >> 1) * 16;
        const __bf16* Bsrc = mat ? Ut : Wt;
        const __bf16* Asrc = mat ? AH : AX;
        float*        Zdst = mat ? ZU : ZW;

        v8f acc0 = {};   // M-subtile 0 (rows 0..15)
        v8f acc1 = {};   // M-subtile 1 (rows 16..31)
#pragma unroll
        for (int k0 = 0; k0 < DK; k0 += 32) {
            // B fragment: two 8-element chunks at [n0+lrow][k0 + hi*8] and +16
            v16bf bf = load_a_frag(Bsrc + (size_t)(n0 + lrow) * DK
                                            + k0 + hi * 8);
            v16bf a0 = load_a_frag(Asrc + (lrow)      * DK + k0 + hi * 8);
            v16bf a1 = load_a_frag(Asrc + (16 + lrow) * DK + k0 + hi * 8);
            acc0 = __builtin_amdgcn_wmma_f32_16x16x32_bf16(
                false, a0, false, bf, (short)0, acc0, false, false);
            acc1 = __builtin_amdgcn_wmma_f32_16x16x32_bf16(
                false, a1, false, bf, (short)0, acc1, false, false);
        }
        // C/D layout: elem e -> row = e + hi*8, col = n0 + lrow
#pragma unroll
        for (int e = 0; e < 8; ++e) {
            Zdst[(hi * 8 + e) * N4 + n0 + lrow]        = acc0[e];
            Zdst[(16 + hi * 8 + e) * N4 + n0 + lrow]   = acc1[e];
        }
    }
    __syncthreads();

    // ---- LN + gates: each wave owns 4 rows ----
    for (int rr = 0; rr < 4; ++rr) {
        const int r = wave * 4 + rr;

        float mW[4], rW[4], mU[4], rU[4];
#pragma unroll
        for (int g = 0; g < 4; ++g) {
            const float* rw = &ZW[r * N4 + g * 256];
            const float* ru = &ZU[r * N4 + g * 256];
            float s1 = 0.f, s2 = 0.f, t1 = 0.f, t2 = 0.f;
            for (int u = lane; u < 256; u += 32) {
                float a = rw[u]; s1 += a; s2 += a * a;
                float b = ru[u]; t1 += b; t2 += b * b;
            }
            s1 = wave_sum(s1); s2 = wave_sum(s2);
            t1 = wave_sum(t1); t2 = wave_sum(t2);
            float mw = s1 * (1.0f / 256.0f);
            float vw = s2 * (1.0f / 256.0f) - mw * mw;
            float mu = t1 * (1.0f / 256.0f);
            float vu = t2 * (1.0f / 256.0f) - mu * mu;
            mW[g] = mw; rW[g] = 1.0f / (sqrtf(vw + EPS_LN) + EPS_LN);
            mU[g] = mu; rU[g] = 1.0f / (sqrtf(vu + EPS_LN) + EPS_LN);
        }

        const float* crow = cin + (size_t)(row0 + r) * 256;
        float cs1 = 0.f, cs2 = 0.f;
        for (int u = lane; u < 256; u += 32) {
            float zi = g1[u]       * (ZW[r*N4 + u]        - mW[0]) * rW[0] + b1[u]
                     + g2[u]       * (ZU[r*N4 + u]        - mU[0]) * rU[0] + b2[u];
            float zf = g1[256 + u] * (ZW[r*N4 + 256 + u]  - mW[1]) * rW[1] + b1[256 + u]
                     + g2[256 + u] * (ZU[r*N4 + 256 + u]  - mU[1]) * rU[1] + b2[256 + u];
            float zg = g1[512 + u] * (ZW[r*N4 + 512 + u]  - mW[2]) * rW[2] + b1[512 + u]
                     + g2[512 + u] * (ZU[r*N4 + 512 + u]  - mU[2]) * rU[2] + b2[512 + u];
            float ig = sigmoidf(zi);
            float fg = sigmoidf(zf);
            float gg = tanhf(zg);
            float cn = fg * crow[u] + ig * gg;
            CT[r * 256 + u] = cn;
            outC[(size_t)(row0 + r) * 256 + u] = cn;
            cs1 += cn; cs2 += cn * cn;
        }
        cs1 = wave_sum(cs1); cs2 = wave_sum(cs2);
        float mc = cs1 * (1.0f / 256.0f);
        float vc = cs2 * (1.0f / 256.0f) - mc * mc;
        float rc = 1.0f / (sqrtf(vc + EPS_LN) + EPS_LN);

        for (int u = lane; u < 256; u += 32) {
            float zo = g1[768 + u] * (ZW[r*N4 + 768 + u] - mW[3]) * rW[3] + b1[768 + u]
                     + g2[768 + u] * (ZU[r*N4 + 768 + u] - mU[3]) * rU[3] + b2[768 + u];
            float og = sigmoidf(zo);
            float cn = CT[r * 256 + u];
            float lc = g3[u] * (cn - mc) * rc + b3[u];
            outH[(size_t)(row0 + r) * 256 + u] = og * tanhf(lc);
        }
    }
}

// ---------------------------------------------------------------------------
extern "C" void kernel_launch(void* const* d_in, const int* in_sizes, int n_in,
                              void* d_out, int out_size, void* d_ws, size_t ws_size,
                              hipStream_t stream) {
    const float* x  = (const float*)d_in[0];
    const float* h  = (const float*)d_in[1];
    const float* c  = (const float*)d_in[2];
    const float* Kw = (const float*)d_in[3];
    const float* Rw = (const float*)d_in[4];
    const float* g1 = (const float*)d_in[5];
    const float* b1 = (const float*)d_in[6];
    const float* g2 = (const float*)d_in[7];
    const float* b2 = (const float*)d_in[8];
    const float* g3 = (const float*)d_in[9];
    const float* b3 = (const float*)d_in[10];

    const int B = in_sizes[0] / DK;         // 65536
    float* outH = (float*)d_out;
    float* outC = outH + (size_t)B * 256;

    __bf16* Wt = (__bf16*)d_ws;             // 512 KB
    __bf16* Ut = Wt + (size_t)DK * N4;      // 512 KB

    (void)hipFuncSetAttribute((const void*)lstm_fused,
                              hipFuncAttributeMaxDynamicSharedMemorySize,
                              SMEM_BYTES);

    cvt_weights<<<(DK * N4 + 255) / 256, 256, 0, stream>>>(Kw, Rw, Wt, Ut);

    lstm_fused<<<B / M_TILE, NTHREADS, SMEM_BYTES, stream>>>(
        x, h, c, Wt, Ut, g1, b1, g2, b2, g3, b3, outH, outC);
}
